// GCNLinkPredictor_88742614270706
// MI455X (gfx1250) — compile-verified, matrix operand order checked by reading, not run on hardware
//
#include <hip/hip_runtime.h>
#include <hip/hip_bf16.h>

typedef __attribute__((ext_vector_type(16))) __bf16 v16bf;
typedef __attribute__((ext_vector_type(8)))  float  v8f;

#define N_NODES 100000
#define F_OUT   128
#define M_TILE  32       // 100000 % 32 == 0 -> 3125 blocks, no guards
#define KCHUNK  128
#define LDC     136      // padded LDS K-stride (bf16): 272B rows -> 4-bank step

// ---------- gfx1250 async global->LDS copy (ASYNCcnt path), with safe fallback ----------
#if defined(__has_builtin)
#if __has_builtin(__builtin_amdgcn_global_load_async_to_lds_b128)
#define HAVE_ASYNC_LDS 1
#endif
#endif

typedef int v4i_vec __attribute__((vector_size(16)));
typedef __attribute__((address_space(1))) v4i_vec* v4i_gptr;   // global
typedef __attribute__((address_space(3))) v4i_vec* v4i_lptr;   // LDS

__device__ __forceinline__ void copy16_g2l(const __bf16* __restrict__ g, __bf16* l) {
#ifdef HAVE_ASYNC_LDS
    __builtin_amdgcn_global_load_async_to_lds_b128(
        (v4i_gptr)(void*)g, (v4i_lptr)(void*)l, 0, 0);
#else
    *(uint4*)l = *(const uint4*)g;
#endif
}

__device__ __forceinline__ void async_wait_all() {
#ifdef HAVE_ASYNC_LDS
#if __has_builtin(__builtin_amdgcn_s_wait_asynccnt)
    __builtin_amdgcn_s_wait_asynccnt(0);
#else
    asm volatile("s_wait_asynccnt 0" ::: "memory");
#endif
#endif
}

// ---------------- degree / normalization ----------------

__global__ void k_deg_init(float* __restrict__ deg, int n) {
    int i = blockIdx.x * blockDim.x + threadIdx.x;
    if (i < n) deg[i] = 1.0f;                     // self-loop contribution
}

__global__ void k_deg_count(const int* __restrict__ dst, float* __restrict__ deg, int E) {
    int e = blockIdx.x * blockDim.x + threadIdx.x;
    if (e < E) atomicAdd(&deg[dst[e]], 1.0f);
}

__global__ void k_rsqrt(const float* __restrict__ deg, float* __restrict__ dis, int n) {
    int i = blockIdx.x * blockDim.x + threadIdx.x;
    if (i < n) dis[i] = rsqrtf(deg[i]);           // deg >= 1 always
}

// ---------------- W pre-pass: Wt[n][k] = bf16(W[k][n]) ----------------

__global__ void k_w_transpose_bf16(const float* __restrict__ W, __bf16* __restrict__ Wt, int K) {
    int i = blockIdx.x * blockDim.x + threadIdx.x;   // over K*128
    if (i >= K * F_OUT) return;
    int n = i & (F_OUT - 1), k = i >> 7;
    Wt[n * K + k] = (__bf16)W[i];
}

// ---------------- WMMA bf16 GEMM: out[N,128] = A[N,K] @ W[K,128] ----------------
// block = 256 threads = 8 waves. Block owns 32 rows x 128 cols.
// Wave w owns cols [16w,16w+16) with two 16x16 accumulators (row subtiles s=0,1).

template <int ABF16>
__global__ __launch_bounds__(256) void k_gemm_wmma_bf16(
    const void* __restrict__ Ain, const __bf16* __restrict__ Wt,
    float* __restrict__ out, int K)
{
    __shared__ __attribute__((aligned(16))) __bf16 sA [M_TILE * LDC];  // [r][k]
    __shared__ __attribute__((aligned(16))) __bf16 sBt[F_OUT  * LDC];  // [n][k]

    const int tid     = threadIdx.x;
    const int rowBase = blockIdx.x * M_TILE;
    const int lane    = tid & 31;
    const int wave    = tid >> 5;
    const int m       = lane & 15;
    const int half    = lane >> 4;     // A/B layout: half 0 -> K {0..7,16..23}, half 1 -> {8..15,24..31}
    const int col0    = wave * 16;

    v8f c0 = {}, c1 = {};
    for (int kc = 0; kc < K; kc += KCHUNK) {
        __syncthreads();                                    // WAR: LDS consumers done
        if (ABF16) {
            // A already bf16: straight 16B-chunk copies (async LDS path)
            const __bf16* Ab = (const __bf16*)Ain;
            for (int cch = tid; cch < M_TILE * 16; cch += 256) {
                int r = cch >> 4, k = (cch & 15) * 8;
                copy16_g2l(&Ab[(size_t)(rowBase + r) * K + kc + k], &sA[r * LDC + k]);
            }
        } else {
            // fp32 A: float4 loads -> 8x bf16 -> one b128 LDS store
            const float* Af = (const float*)Ain;
            for (int i = tid * 8; i < M_TILE * KCHUNK; i += 256 * 8) {
                int r = i >> 7, k = i & 127;
                const float4* src = (const float4*)&Af[(size_t)(rowBase + r) * K + kc + k];
                float4 f0 = src[0], f1 = src[1];
                union { __bf16 h[8]; uint4 u; } p;
                p.h[0] = (__bf16)f0.x; p.h[1] = (__bf16)f0.y;
                p.h[2] = (__bf16)f0.z; p.h[3] = (__bf16)f0.w;
                p.h[4] = (__bf16)f1.x; p.h[5] = (__bf16)f1.y;
                p.h[6] = (__bf16)f1.z; p.h[7] = (__bf16)f1.w;
                *(uint4*)&sA[r * LDC + k] = p.u;
            }
        }
        // W^T chunk: straight 16B-chunk copies (async LDS path)
        for (int cch = tid; cch < F_OUT * 16; cch += 256) {
            int n = cch >> 4, k = (cch & 15) * 8;
            copy16_g2l(&Wt[(size_t)n * K + kc + k], &sBt[n * LDC + k]);
        }
        async_wait_all();
        __syncthreads();                                    // RAW: staging visible

        for (int kb = 0; kb < KCHUNK; kb += 32) {
            const int k0 = kb + half * 8;
            v16bf a0, a1, b;
#pragma unroll
            for (int t = 0; t < 8; ++t) {
                a0[t]     = sA[m * LDC + k0 + t];
                a0[t + 8] = sA[m * LDC + k0 + 16 + t];
                a1[t]     = sA[(16 + m) * LDC + k0 + t];
                a1[t + 8] = sA[(16 + m) * LDC + k0 + 16 + t];
                b[t]      = sBt[(col0 + m) * LDC + k0 + t];
                b[t + 8]  = sBt[(col0 + m) * LDC + k0 + 16 + t];
            }
            c0 = __builtin_amdgcn_wmma_f32_16x16x32_bf16(false, a0, false, b, (short)0, c0, false, false);
            c1 = __builtin_amdgcn_wmma_f32_16x16x32_bf16(false, a1, false, b, (short)0, c1, false, false);
        }
    }
    // C layout: VGPR r -> row (r + 8*half), col = lane%16
#pragma unroll
    for (int r = 0; r < 8; ++r) {
        int row = rowBase + half * 8 + r;
        out[(size_t)row        * F_OUT + col0 + m] = c0[r];
        out[(size_t)(row + 16) * F_OUT + col0 + m] = c1[r];
    }
}

// ---------------- aggregation ----------------

// out[i,:] = dis[i]^2 * h[i,:] + bias   (self-loop + bias folded), float4-vectorized
__global__ void k_agg_init(const float4* __restrict__ h, const float* __restrict__ dis,
                           const float4* __restrict__ bias, float4* __restrict__ out, int total4)
{
    int i = blockIdx.x * blockDim.x + threadIdx.x;
    if (i >= total4) return;
    int node = i >> 5, j = i & 31;                 // 32 float4 per row of 128
    float d = dis[node];
    float d2 = d * d;
    float4 hv = h[i], bv = bias[j];
    out[i] = make_float4(d2 * hv.x + bv.x, d2 * hv.y + bv.y,
                         d2 * hv.z + bv.z, d2 * hv.w + bv.w);
}

// one wave32 per edge: gather h[src] (one 512B line), scale, 4 f32 atomics/lane into out[dst]
__global__ __launch_bounds__(256) void k_edge_agg(
    const float* __restrict__ h, const int* __restrict__ src, const int* __restrict__ dst,
    const float* __restrict__ dis, float* __restrict__ out, int E)
{
    int e    = __builtin_amdgcn_readfirstlane((blockIdx.x * blockDim.x + threadIdx.x) >> 5);
    int lane = threadIdx.x & 31;
    if (e >= E) return;
    int s = src[e], d = dst[e];                    // uniform -> scalar loads
    float nrm = dis[s] * dis[d];
    const float4* hs = (const float4*)(h + (size_t)s * F_OUT);
    float*        od = out + (size_t)d * F_OUT + lane * 4;
    float4 msg = hs[lane];
    atomicAdd(od + 0, nrm * msg.x);
    atomicAdd(od + 1, nrm * msg.y);
    atomicAdd(od + 2, nrm * msg.z);
    atomicAdd(od + 3, nrm * msg.w);
}

// fused relu + fp32->bf16: h1bf = bf16(max(a,0)); layer-2 GEMM reads half the bytes
__global__ void k_relu_bf16(const float4* __restrict__ a, __bf16* __restrict__ o, int total4) {
    int i = blockIdx.x * blockDim.x + threadIdx.x;
    if (i >= total4) return;
    float4 v = a[i];
    union { __bf16 h[4]; uint2 u; } p;
    p.h[0] = (__bf16)fmaxf(v.x, 0.0f);
    p.h[1] = (__bf16)fmaxf(v.y, 0.0f);
    p.h[2] = (__bf16)fmaxf(v.z, 0.0f);
    p.h[3] = (__bf16)fmaxf(v.w, 0.0f);
    *(uint2*)&o[(size_t)i * 4] = p.u;
}

// ---------------- decode: one wave32 per candidate pair ----------------

__global__ __launch_bounds__(256) void k_decode(
    const float* __restrict__ z, const int* __restrict__ u, const int* __restrict__ v,
    float* __restrict__ out, int P)
{
    int p    = __builtin_amdgcn_readfirstlane((blockIdx.x * blockDim.x + threadIdx.x) >> 5);
    int lane = threadIdx.x & 31;
    if (p >= P) return;
    const float4* zu = (const float4*)(z + (size_t)u[p] * F_OUT);
    const float4* zv = (const float4*)(z + (size_t)v[p] * F_OUT);
    float4 a = zu[lane], b = zv[lane];
    float s = a.x * b.x + a.y * b.y + a.z * b.z + a.w * b.w;
#pragma unroll
    for (int off = 16; off > 0; off >>= 1)
        s += __shfl_xor(s, off, 32);
    if (lane == 0) out[p] = s;
}

// ---------------- driver ----------------

extern "C" void kernel_launch(void* const* d_in, const int* in_sizes, int n_in,
                              void* d_out, int out_size, void* d_ws, size_t ws_size,
                              hipStream_t stream)
{
    const float* x  = (const float*)d_in[0];
    const int*   ei = (const int*)  d_in[1];   // [2,E]
    const int*   ep = (const int*)  d_in[2];   // [2,P]
    const float* W1 = (const float*)d_in[3];
    const float* b1 = (const float*)d_in[4];
    const float* W2 = (const float*)d_in[5];
    const float* b2 = (const float*)d_in[6];

    const int N  = N_NODES;
    const int E  = in_sizes[1] / 2;
    const int P  = in_sizes[2] / 2;
    const int K1 = in_sizes[0] / N;            // 256
    const int K2 = F_OUT;                      // 128

    const int* src = ei;
    const int* dst = ei + E;
    const int* pu  = ep;
    const int* pv  = ep + P;

    char* ws = (char*)d_ws;
    float*  B0   = (float*)ws;                          ws += (size_t)N * F_OUT * 4;
    float*  B1   = (float*)ws;                          ws += (size_t)N * F_OUT * 4;
    __bf16* h1bf = (__bf16*)ws;                         ws += (size_t)N * F_OUT * 2;
    __bf16* Wt1  = (__bf16*)ws;                         ws += (size_t)F_OUT * K1 * 2;
    __bf16* Wt2  = (__bf16*)ws;                         ws += (size_t)F_OUT * K2 * 2;
    float*  deg  = (float*)ws;                          ws += (size_t)N * 4;
    float*  dis  = (float*)ws;
    float*  scores = (float*)d_out;

    const int elems  = N * F_OUT;
    const int total4 = elems / 4;
    dim3 blk(256);

    // normalization + weight pre-transpose (independent, cheap)
    k_deg_init       <<<(N + 255) / 256, blk, 0, stream>>>(deg, N);
    k_deg_count      <<<(E + 255) / 256, blk, 0, stream>>>(dst, deg, E);
    k_rsqrt          <<<(N + 255) / 256, blk, 0, stream>>>(deg, dis, N);
    k_w_transpose_bf16<<<(K1 * F_OUT + 255) / 256, blk, 0, stream>>>(W1, Wt1, K1);
    k_w_transpose_bf16<<<(K2 * F_OUT + 255) / 256, blk, 0, stream>>>(W2, Wt2, K2);

    // layer 1: B0 = x @ W1 ; B1 = aggregate(B0) + b1 ; h1bf = bf16(relu(B1))
    k_gemm_wmma_bf16<0><<<N / M_TILE, blk, 0, stream>>>(x, Wt1, B0, K1);
    k_agg_init <<<(total4 + 255) / 256, blk, 0, stream>>>((const float4*)B0, dis, (const float4*)b1, (float4*)B1, total4);
    k_edge_agg <<<(E + 7) / 8, blk, 0, stream>>>(B0, src, dst, dis, B1, E);
    k_relu_bf16<<<(total4 + 255) / 256, blk, 0, stream>>>((const float4*)B1, h1bf, total4);

    // layer 2: B0 = h1 @ W2 ; z(B1) = aggregate(B0) + b2
    k_gemm_wmma_bf16<1><<<N / M_TILE, blk, 0, stream>>>(h1bf, Wt2, B0, K2);
    k_agg_init <<<(total4 + 255) / 256, blk, 0, stream>>>((const float4*)B0, dis, (const float4*)b2, (float4*)B1, total4);
    k_edge_agg <<<(E + 7) / 8, blk, 0, stream>>>(B0, src, dst, dis, B1, E);

    // decode
    k_decode   <<<(P + 7) / 8, blk, 0, stream>>>(B1, pu, pv, scores, P);
}
